// LutLinear_24369644437526
// MI455X (gfx1250) — matrix operand order repaired and checked by reading, not compiled, not run on hardware
//
#include <hip/hip_runtime.h>

typedef __attribute__((ext_vector_type(16))) _Float16 v16h;
typedef __attribute__((ext_vector_type(8)))  float    v8f;

constexpr int Kd = 4096, Nd = 4096, WBIT = 3, NG = 32, WORDS = Kd / 32;

__device__ __forceinline__ unsigned f16b(float v) {
  _Float16 h = (_Float16)v;
  return (unsigned)__builtin_bit_cast(unsigned short, h);
}

// Grid: N/16 blocks x 256 threads (8 waves).
//   waves 0-5: (plane p = w>>1, k-half = w&1): 16 groups x 4 chunks via WMMA
//              A = +/-alpha (f16), B = x in column 0, C accumulates f32.
//   waves 6-7: q_bias * group-sum path, 16 groups each.
// Partials (8 waves x 16 columns) summed in LDS; deterministic, no atomics.
__global__ __launch_bounds__(256) void bcq_gemv_wmma(
    const float* __restrict__ x, const int* __restrict__ bw,
    const float* __restrict__ alpha, const float* __restrict__ qb,
    const float* __restrict__ bias, float* __restrict__ out) {
  __shared__ unsigned xh[Kd / 2];   // x as packed f16 pairs
  __shared__ float    psum[256];    // per-thread partials for group sums
  __shared__ float    sgrp[NG];     // group sums of x (f32)
  __shared__ float    outp[128];    // per-wave partial outputs (8 x 16)

  const int tid = threadIdx.x;
  const int n0  = blockIdx.x * 16;

  // Stage x -> LDS as f16 pairs; 16 values per thread + partial sums.
  {
    const int kbase = tid * 16;
    float s = 0.f;
#pragma unroll
    for (int j = 0; j < 8; ++j) {
      float x0 = x[kbase + 2 * j], x1 = x[kbase + 2 * j + 1];
      s += x0 + x1;
      xh[tid * 8 + j] = f16b(x0) | (f16b(x1) << 16);
    }
    psum[tid] = s;
  }
  __syncthreads();
  if (tid < NG) {
    float s = 0.f;
#pragma unroll
    for (int i = 0; i < 8; ++i) s += psum[tid * 8 + i];
    sgrp[tid] = s;
  }
  __syncthreads();

  const int wid  = tid >> 5;
  const int lane = tid & 31;

  if (wid < 6) {
    const int  p    = wid >> 1;            // bit-plane
    const int  g0   = (wid & 1) * 16;      // first group of this wave's k-half
    const int  ncol = lane & 15;           // output column within tile
    const int  n    = n0 + ncol;
    const int  hi   = (lane >> 4) << 3;    // 0 or 8: lane-half K offset
    const bool bact = (ncol == 0);         // lanes 0 and 16 carry x in B col 0

    union { v16h v; unsigned u[8]; } A, B;
    v8f acc = {};

    for (int gi = 0; gi < 16; ++gi) {
      const int g     = g0 + gi;
      const int cbase = g * 4;
      // one alpha per group, duplicated into an f16x2 word
      const unsigned a2 = f16b(alpha[(g * WBIT + p) * Nd + n]) * 0x00010001u;

      unsigned w[4];
#pragma unroll
      for (int j = 0; j < 4; ++j) {
        w[j] = (unsigned)bw[((cbase + j) * WBIT + p) * Nd + n];
        if (gi + 2 < 16)  // stream-ahead: group g+2 -> global_prefetch_b8
          __builtin_prefetch(&bw[((cbase + 8 + j) * WBIT + p) * Nd + n], 0, 1);
      }

#pragma unroll
      for (int j = 0; j < 4; ++j) {
        const int c = cbase + j;
        unsigned nw = ~w[j];  // clear bit -> flip sign (xor 0x8000 per half)
        // A 16x32 f16: lanes<16 hold K 0..15; lanes>=16 K 16..31
#pragma unroll
        for (int i = 0; i < 8; ++i) {
          int pp = 2 * hi + 2 * i;
          unsigned m = (((nw >> pp) & 1u) << 15) | (((nw >> (pp + 1)) & 1u) << 31);
          A.u[i] = a2 ^ m;
        }
        // B 32x16 f16: column 0 = x chunk; lane0 K=0..15, lane16 K=16..31
#pragma unroll
        for (int i = 0; i < 8; ++i)
          B.u[i] = bact ? xh[c * 16 + hi + i] : 0u;

        acc = __builtin_amdgcn_wmma_f32_16x16x32_f16(
            false, A.v, false, B.v, (short)0, acc, false, false);
      }
    }

    // C col 0: lane 0 holds rows 0-7 (VGPRs 0-7), lane 16 rows 8-15.
    if (lane == 0) {
#pragma unroll
      for (int v = 0; v < 8; ++v) outp[wid * 16 + v] = acc[v];
    }
    if (lane == 16) {
#pragma unroll
      for (int v = 0; v < 8; ++v) outp[wid * 16 + 8 + v] = acc[v];
    }
  } else if (lane < 16) {
    // q_bias path: out_q[n] = sum_g q_bias[g,n] * s[g]; wave6 g=0..15, wave7 g=16..31
    const int n  = n0 + lane;
    const int gb = (wid - 6) * 16;
    float a = 0.f;
#pragma unroll 4
    for (int g = gb; g < gb + 16; ++g) a += qb[g * Nd + n] * sgrp[g];
    outp[wid * 16 + lane] = a;
  }
  __syncthreads();

  if (tid < 16) {
    float r = bias[n0 + tid];
#pragma unroll
    for (int w8 = 0; w8 < 8; ++w8) r += outp[w8 * 16 + tid];
    out[n0 + tid] = r;
  }
}

extern "C" void kernel_launch(void* const* d_in, const int* in_sizes, int n_in,
                              void* d_out, int out_size, void* d_ws, size_t ws_size,
                              hipStream_t stream) {
  (void)in_sizes; (void)n_in; (void)d_ws; (void)ws_size; (void)out_size;
  const float* x     = (const float*)d_in[0];
  const int*   bw    = (const int*)d_in[1];
  const float* alpha = (const float*)d_in[2];
  const float* qbias = (const float*)d_in[3];
  const float* bias  = (const float*)d_in[4];
  float*       out   = (float*)d_out;

  bcq_gemv_wmma<<<Nd / 16, 256, 0, stream>>>(x, bw, alpha, qbias, bias, out);
}
